// FLUKEModel_45921790329437
// MI455X (gfx1250) — matrix operand order repaired and checked by reading, not compiled, not run on hardware
//
#include <hip/hip_runtime.h>
#include <math.h>

// ---------------------------------------------------------------------------
// FLUKE late-interaction scorer for MI455X (gfx1250, wave32, WMMA)
//   B=16, Nq=32, Ndocs=256, Nd=180, D=128, HID=64, TOPK=3, TEMP=0.1
// ---------------------------------------------------------------------------

#define EMBED_DIM 128
#define NQ        32
#define NB        16
#define NDOCS     256
#define ND        180
#define NDPAD     192          // 12 k-tiles of 16
#define NKT       12
#define HID       64
#define NEGVAL    (-1000000000.0f)
#define BIGNEG    (-4.0e9f)
#define INV_TEMP  10.0f

typedef __attribute__((ext_vector_type(16))) _Float16 v16h;
typedef __attribute__((ext_vector_type(8)))  _Float16 v8h;
typedef __attribute__((ext_vector_type(8)))  float    v8f;

union V16U { v16h v; v8h h[2]; };

// branchless sorted-insert into (t0 >= t1 >= t2)
__device__ __forceinline__ void top3_insert(float v, float& t0, float& t1, float& t2) {
  float m0 = fminf(t0, v);
  t0 = fmaxf(t0, v);
  float m1 = fminf(t1, m0);
  t1 = fmaxf(t1, m0);
  t2 = fmaxf(t2, m1);
}

// ---------------------------------------------------------------------------
// Kernel A: per-query importance (CLS attention + GELU MLP + softmax * nvalid)
// One block per batch b; tiny VALU work.
// ---------------------------------------------------------------------------
__global__ __launch_bounds__(128) void cqi_kernel(
    const float* __restrict__ q,      // (16,32,128)
    const float* __restrict__ Wp,     // (128,128)
    const float* __restrict__ bp,     // (128)
    const float* __restrict__ W1,     // (64,128)
    const float* __restrict__ b1,     // (64)
    const float* __restrict__ W2,     // (1,64)
    const float* __restrict__ b2,     // (1)
    const unsigned char* __restrict__ qmask, // (16,32) bool
    float* __restrict__ imp)          // (16,32) out
{
  __shared__ float sCls[EMBED_DIM];
  __shared__ float sProj[EMBED_DIM];
  __shared__ float sRaw[NQ];

  const int b = blockIdx.x;
  const int t = threadIdx.x;

  if (t < EMBED_DIM) sCls[t] = q[(b * NQ) * EMBED_DIM + t];
  __syncthreads();

  if (t < EMBED_DIM) {                 // proj = cls @ Wp^T + bp
    float acc = bp[t];
    const float* w = Wp + t * EMBED_DIM;
    #pragma unroll 4
    for (int d0 = 0; d0 < EMBED_DIM; ++d0) acc = fmaf(w[d0], sCls[d0], acc);
    sProj[t] = acc;
  }
  __syncthreads();

  if (t < NQ) {
    const float* qrow = q + (b * NQ + t) * EMBED_DIM;
    float attn = 0.f;
    #pragma unroll 4
    for (int d0 = 0; d0 < EMBED_DIM; ++d0) attn = fmaf(sProj[d0], qrow[d0], attn);
    float tok = b2[0];
    for (int h = 0; h < HID; ++h) {    // gelu(q@W1^T+b1) @ W2^T + b2
      float acc = b1[h];
      const float* w = W1 + h * EMBED_DIM;
      #pragma unroll 4
      for (int d0 = 0; d0 < EMBED_DIM; ++d0) acc = fmaf(w[d0], qrow[d0], acc);
      float g = 0.5f * acc * (1.0f + erff(acc * 0.70710678118654752f)); // exact gelu
      tok = fmaf(W2[h], g, tok);
    }
    sRaw[t] = qmask[b * NQ + t] ? (attn + tok) : NEGVAL;
  }
  __syncthreads();

  if (t < NQ) {                        // softmax over q, scaled by nvalid
    float mx = -3.0e38f; int nv = 0;
    for (int i = 0; i < NQ; ++i) { mx = fmaxf(mx, sRaw[i]); nv += qmask[b * NQ + i] ? 1 : 0; }
    float se = 0.f;
    for (int i = 0; i < NQ; ++i) se += __expf(sRaw[i] - mx);
    imp[b * NQ + t] = __expf(sRaw[t] - mx) / se * (float)nv;
  }
}

// ---------------------------------------------------------------------------
// Kernel B: fused sim-GEMM (WMMA f16) + top-3 + temp-softmax + q-reduction.
// Grid: 256 blocks (one per doc n), 128 threads = 4 waves.
// Doc tile staged to LDS ONCE; two wave-pairs sweep even/odd batches.
// simT tiles: A = doc tokens (M=k), B = Q^T (N=q); with the CDNA5 C layout
// each lane owns 8 k-values for a single q per tile -> register top-3,
// one shfl_xor(16) merge, then a wave reduction over q.
// ---------------------------------------------------------------------------
__global__ __launch_bounds__(128) void fluke_kernel(
    const float* __restrict__ q,      // (16,32,128)
    const float* __restrict__ d,      // (256,180,128)
    const float* __restrict__ imp,    // (16,32)
    const unsigned char* __restrict__ qmask, // (16,32)
    const unsigned char* __restrict__ dmask, // (256,180)
    float* __restrict__ out)          // (16,256)
{
  __shared__ _Float16 sD[ND * EMBED_DIM];      // 180*128*2 = 46080 B
  __shared__ float    sBias[NDPAD];            // 0 if valid k, -4e9 otherwise
  __shared__ _Float16 sQ[2][NQ * EMBED_DIM];   // per-pair Q block, 2*8 KB
  __shared__ float    sPartial[2][2];

  const int n   = blockIdx.x;
  const int tid = threadIdx.x;

  // --- stage doc tile (f32 -> f16) + mask bias, once per block ---
  {
    const float* dsrc = d + (size_t)n * ND * EMBED_DIM;
    for (int i = tid; i < ND * EMBED_DIM; i += 128) sD[i] = (_Float16)dsrc[i];
    for (int k = tid; k < NDPAD; k += 128)
      sBias[k] = (k < ND && dmask[n * ND + k]) ? 0.0f : BIGNEG;
  }

  const int pair = tid >> 6;        // 0: even b, 1: odd b
  const int tp   = tid & 63;        // thread within pair
  const int wip  = (tid >> 5) & 1;  // wave within pair -> q-tile
  const int lane = tid & 31;
  const int half = lane >> 4;
  const int m    = lane & 15;
  const int qrow = wip * 16 + m;    // q index 0..31

  for (int bi = 0; bi < NB / 2; ++bi) {
    const int b = bi * 2 + pair;

    // stage this pair's Q block (f32 -> f16)
    const float* qsrc = q + (size_t)b * NQ * EMBED_DIM;
    for (int i = tp; i < NQ * EMBED_DIM; i += 64) sQ[pair][i] = (_Float16)qsrc[i];
    if (bi + 1 < NB / 2) {          // prefetch next Q block into cache
      const float* nxt = q + (size_t)(b + 2) * NQ * EMBED_DIM;
      __builtin_prefetch(nxt + tp * 64, 0, 3);
    }
    __syncthreads();

    float t0 = NEGVAL, t1 = NEGVAL, t2 = NEGVAL;   // running top-3 (desc)

    for (int kt = 0; kt < NKT; ++kt) {
      v8f c = {};
      int arow = kt * 16 + m;                      // doc-token row for A fragment
      arow = arow < ND ? arow : (ND - 1);          // clamp pad rows (bias kills them)
      #pragma unroll
      for (int kk = 0; kk < EMBED_DIM; kk += 32) {
        // A (16x32 f16): elems 0-7 -> K = kk+8*half.. ; 8-15 -> K = kk+16+8*half..
        V16U a;
        a.h[0] = *(const v8h*)&sD[arow * EMBED_DIM + kk + 8 * half];
        a.h[1] = *(const v8h*)&sD[arow * EMBED_DIM + kk + 16 + 8 * half];
        // B (32x16 f16): N = m, elems 0-7 -> K = kk+8*half.. ; 8-15 -> K = kk+16+8*half..
        V16U bm;
        bm.h[0] = *(const v8h*)&sQ[pair][qrow * EMBED_DIM + kk + 8 * half];
        bm.h[1] = *(const v8h*)&sQ[pair][qrow * EMBED_DIM + kk + 16 + 8 * half];
        c = __builtin_amdgcn_wmma_f32_16x16x32_f16(false, a.v, false, bm.v,
                                                   (short)0, c, false, false);
      }
      // C: VGPR r -> k = kt*16 + 8*half + r, all for q = qrow
      const v8f bias = *(const v8f*)&sBias[kt * 16 + 8 * half];
      #pragma unroll
      for (int r = 0; r < 8; ++r) {
        const float v = fmaxf(c[r] + bias[r], NEGVAL);   // masked/pad -> exactly NEG
        top3_insert(v, t0, t1, t2);
      }
    }

    // merge top-3 across the two lane-halves sharing the same q (branchless)
    {
      float o0 = __shfl_xor(t0, 16, 32);
      float o1 = __shfl_xor(t1, 16, 32);
      float o2 = __shfl_xor(t2, 16, 32);
      top3_insert(o0, t0, t1, t2);
      top3_insert(o1, t0, t1, t2);
      top3_insert(o2, t0, t1, t2);
    }

    // softmax(top3 / TEMP) . top3
    const float e1 = __expf((t1 - t0) * INV_TEMP);
    const float e2 = __expf((t2 - t0) * INV_TEMP);
    float tok = (t0 + e1 * t1 + e2 * t2) / (1.0f + e1 + e2);

    const unsigned char qm = qmask[b * NQ + qrow];
    float contrib = (half == 0 && qm) ? tok * imp[b * NQ + qrow] : 0.0f;

    // wave reduction over the 16 q's (half==1 lanes contribute zero)
    #pragma unroll
    for (int off = 16; off > 0; off >>= 1) contrib += __shfl_xor(contrib, off, 32);

    if (lane == 0) sPartial[pair][wip] = contrib;
    __syncthreads();
    if (tp == 0) out[b * NDOCS + n] = sPartial[pair][0] + sPartial[pair][1];
    // next iteration's Q staging is safe: all reads of sQ[pair] completed above
  }
}

// ---------------------------------------------------------------------------
extern "C" void kernel_launch(void* const* d_in, const int* in_sizes, int n_in,
                              void* d_out, int out_size, void* d_ws, size_t ws_size,
                              hipStream_t stream) {
  const float* q    = (const float*)d_in[0];
  const float* dd   = (const float*)d_in[1];
  const float* Wp   = (const float*)d_in[2];
  const float* bp   = (const float*)d_in[3];
  const float* W1   = (const float*)d_in[4];
  const float* b1   = (const float*)d_in[5];
  const float* W2   = (const float*)d_in[6];
  const float* b2   = (const float*)d_in[7];
  const unsigned char* qmask = (const unsigned char*)d_in[8];
  const unsigned char* dmask = (const unsigned char*)d_in[9];
  float* out = (float*)d_out;
  float* imp = (float*)d_ws;            // 16*32 floats of scratch

  cqi_kernel<<<dim3(16), dim3(128), 0, stream>>>(q, Wp, bp, W1, b1, W2, b2, qmask, imp);

  fluke_kernel<<<dim3(NDOCS), dim3(128), 0, stream>>>(q, dd, imp, qmask, dmask, out);
}
